// Linear_4114578669651
// MI455X (gfx1250) — matrix operand order repaired and checked by reading, not compiled, not run on hardware
//
#include <hip/hip_runtime.h>

typedef __attribute__((ext_vector_type(16))) __bf16 v16bf;
typedef __attribute__((ext_vector_type(8)))  __bf16 v8bf;
typedef __attribute__((ext_vector_type(8)))  float  v8f;
typedef __attribute__((ext_vector_type(4)))  float  v4f;

#define M_TOT 4096
#define N_TOT 8192
#define K_TOT 2048
#define LDW   (K_TOT + 1)        // 2049 floats per matrix row (weights + bias)

#define BM 128
#define BN 256
#define BK 64
#define LDST 72                  // bf16 per LDS row: 144B = 9*16B -> aligned b128, conflict-free
#define A_ELEMS (BM * LDST)
#define B_ELEMS (BN * LDST)
#define BUF_ELEMS (A_ELEMS + B_ELEMS)

union FragBF16 { v16bf v; v8bf h[2]; };

// ---------------- pack pass: f32 -> bf16 once, bias extracted ----------------
__global__ __launch_bounds__(256) void pack_x_bf16(const float* __restrict__ X,
                                                   __bf16* __restrict__ Xbf) {
    size_t i = ((size_t)blockIdx.x * 256 + threadIdx.x) * 8;
    v4f a = *(const v4f*)(X + i);
    v4f b = *(const v4f*)(X + i + 4);
    v8bf o;
    o[0] = (__bf16)a.x; o[1] = (__bf16)a.y; o[2] = (__bf16)a.z; o[3] = (__bf16)a.w;
    o[4] = (__bf16)b.x; o[5] = (__bf16)b.y; o[6] = (__bf16)b.z; o[7] = (__bf16)b.w;
    *(v8bf*)(Xbf + i) = o;
}

__global__ __launch_bounds__(256) void pack_w_bf16(const float* __restrict__ Wm,
                                                   __bf16* __restrict__ Wbf,
                                                   float* __restrict__ bias) {
    size_t i = (size_t)blockIdx.x * 256 + threadIdx.x;   // one 8-elem chunk each
    int row = (int)(i >> 8);            // 256 chunks per 2048-wide row
    int c   = ((int)i & 255) * 8;
    const float* src = Wm + (size_t)row * LDW + c;       // rows only 4B-aligned -> scalar loads
    v8bf o;
    #pragma unroll
    for (int j = 0; j < 8; ++j) o[j] = (__bf16)src[j];
    *(v8bf*)(Wbf + (size_t)row * K_TOT + c) = o;
    if (c == 0) bias[row] = Wm[(size_t)row * LDW + K_TOT];
}

// -------- async global->LDS DMA (CDNA5, ASYNCcnt-tracked, no VGPR round-trip) --------
__device__ __forceinline__ void async_b128(unsigned lds_byte, unsigned g_byte_off,
                                           const __bf16* gbase) {
    asm volatile("global_load_async_to_lds_b128 %0, %1, %2"
                 :: "v"(lds_byte), "v"(g_byte_off), "s"(gbase) : "memory");
}
__device__ __forceinline__ void wait_async0() {
    asm volatile("s_wait_asynccnt 0" ::: "memory");
}

// ---------------- main GEMM: Y = Xbf * Wbf^T + bias, f32 accumulate ----------------
// Block 128x256, 8 waves in 2(M) x 4(N); each wave owns a 64x64 patch = 4x4 16x16 tiles.
__global__ __launch_bounds__(256) void linear_wmma_bf16(
    const __bf16* __restrict__ Xbf,   // [4096][2048]
    const __bf16* __restrict__ Wbf,   // [8192][2048]
    const float*  __restrict__ bias,  // [8192]
    float* __restrict__ Y)            // [4096][8192]
{
    __shared__ __bf16 lds[2 * BUF_ELEMS];   // ping-pong: {A[128][72], B[256][72]} x2 = 108KB

    const int tid  = threadIdx.x;
    const int lane = tid & 31;
    const int wave = tid >> 5;          // 0..7
    const int wm   = wave >> 2;         // 0..1 (M)
    const int wn   = wave & 3;          // 0..3 (N)
    const int half = lane >> 4;
    const int l16  = lane & 15;
    const int bm   = blockIdx.y * BM;
    const int bn   = blockIdx.x * BN;

    // staging geometry: 8-bf16 (16B) chunks. A: 1024 chunks (4/thread), B: 2048 (8/thread)
    const int srow = tid >> 3;          // 0..31 (+32q)
    const int scol = (tid & 7) * 8;     // bf16 column 0..56

    unsigned aLds[4], aG[4], bLds[8], bG[8];
    #pragma unroll
    for (int q = 0; q < 4; ++q) {
        const int r = srow + 32 * q;
        aLds[q] = (unsigned)(uintptr_t)&lds[r * LDST + scol];
        aG[q]   = (unsigned)(((bm + r) * K_TOT + scol) * 2);   // byte offsets
    }
    #pragma unroll
    for (int q = 0; q < 8; ++q) {
        const int r = srow + 32 * q;    // 0..255
        bLds[q] = (unsigned)(uintptr_t)&lds[A_ELEMS + r * LDST + scol];
        bG[q]   = (unsigned)(((bn + r) * K_TOT + scol) * 2);
    }

    auto stage = [&](int s, int kt) {
        const unsigned bufo = (unsigned)(s * BUF_ELEMS * 2);
        const unsigned ko   = (unsigned)(kt * 2);
        #pragma unroll
        for (int q = 0; q < 4; ++q) async_b128(aLds[q] + bufo, aG[q] + ko, Xbf);
        #pragma unroll
        for (int q = 0; q < 8; ++q) async_b128(bLds[q] + bufo, bG[q] + ko, Wbf);
    };

    v8f acc[4][4];
    #pragma unroll
    for (int i = 0; i < 4; ++i)
        #pragma unroll
        for (int j = 0; j < 4; ++j)
            #pragma unroll
            for (int e = 0; e < 8; ++e) acc[i][j][e] = 0.0f;

    stage(0, 0);
    wait_async0();
    __syncthreads();

    int s = 0;
    for (int kt = 0; kt < K_TOT; kt += BK) {
        const bool more = (kt + BK) < K_TOT;
        if (more) stage(s ^ 1, kt + BK);           // DMA next tile while computing this one

        const __bf16* A = &lds[s * BUF_ELEMS];
        const __bf16* B = &lds[s * BUF_ELEMS + A_ELEMS];

        #pragma unroll
        for (int kc = 0; kc < 2; ++kc) {           // two 16x16x32 K-chunks per BK=64
            FragBF16 af[4], bfm[4];
            const int klo = kc * 32 + half * 8;    // A layout: ISA 7.12.2
            #pragma unroll
            for (int mt = 0; mt < 4; ++mt) {
                const int row = wm * 64 + mt * 16 + l16;
                af[mt].h[0] = *(const v8bf*)(&A[row * LDST + klo]);
                af[mt].h[1] = *(const v8bf*)(&A[row * LDST + klo + 16]);
            }
            const int kb = kc * 32 + half * 8;     // B layout: same K order as A (ISA 7.12.2)
            #pragma unroll
            for (int nt = 0; nt < 4; ++nt) {
                const int col = wn * 64 + nt * 16 + l16;
                bfm[nt].h[0] = *(const v8bf*)(&B[col * LDST + kb]);
                bfm[nt].h[1] = *(const v8bf*)(&B[col * LDST + kb + 16]);
            }
            #pragma unroll
            for (int mt = 0; mt < 4; ++mt)
                #pragma unroll
                for (int nt = 0; nt < 4; ++nt)
                    acc[mt][nt] = __builtin_amdgcn_wmma_f32_16x16x32_bf16(
                        false, af[mt].v, false, bfm[nt].v,
                        (short)0, acc[mt][nt], false, false);
        }

        if (more) wait_async0();                    // next tile landed in LDS
        __syncthreads();                            // everyone done reading s / writing s^1
        s ^= 1;
    }

    // ---- epilogue: + bias, nontemporal f32 stores (output is 128MB streamed) ----
    #pragma unroll
    for (int nt = 0; nt < 4; ++nt) {
        const int N = bn + wn * 64 + nt * 16 + l16;
        const float bz = bias[N];
        #pragma unroll
        for (int mt = 0; mt < 4; ++mt) {
            #pragma unroll
            for (int v = 0; v < 8; ++v) {
                const int Mrow = bm + wm * 64 + mt * 16 + v + half * 8;
                __builtin_nontemporal_store(acc[mt][nt][v] + bz,
                                            &Y[(size_t)Mrow * N_TOT + N]);
            }
        }
    }
}

extern "C" void kernel_launch(void* const* d_in, const int* in_sizes, int n_in,
                              void* d_out, int out_size, void* d_ws, size_t ws_size,
                              hipStream_t stream) {
    const float* X  = (const float*)d_in[0];   // [4096, 2048] f32
    const float* Wm = (const float*)d_in[1];   // [8192, 2049] f32
    float* Y = (float*)d_out;                  // [4096, 8192] f32
    (void)in_sizes; (void)n_in; (void)out_size; (void)ws_size;

    char* ws = (char*)d_ws;                    // layout: Xbf(16MB) | Wbf(32MB) | bias(32KB)
    __bf16* Xbf = (__bf16*)ws;
    __bf16* Wbf = (__bf16*)(ws + (size_t)M_TOT * K_TOT * 2);
    float*  bz  = (float*)(ws + (size_t)M_TOT * K_TOT * 2 + (size_t)N_TOT * K_TOT * 2);

    pack_x_bf16<<<dim3((M_TOT * (size_t)K_TOT / 8) / 256), dim3(256), 0, stream>>>(X, Xbf);
    pack_w_bf16<<<dim3((N_TOT * 256) / 256), dim3(256), 0, stream>>>(Wm, Wbf, bz);

    dim3 grid(N_TOT / BN, M_TOT / BM);         // 32 x 32
    linear_wmma_bf16<<<grid, dim3(256), 0, stream>>>(Xbf, Wbf, bz, Y);
}
